// ChannelAttention_46918222741524
// MI455X (gfx1250) — matrix operand order repaired and checked by reading, not compiled, not run on hardware
//
#include <hip/hip_runtime.h>
#include <stdint.h>

// ---------- CDNA5 WMMA plumbing (wave32, 16x16x32 bf16 -> f32) ----------
typedef __attribute__((ext_vector_type(16))) __bf16        bf16x16;
typedef __attribute__((ext_vector_type(8)))  float         f32x8;
typedef __attribute__((ext_vector_type(8)))  unsigned int  u32x8;
typedef __attribute__((ext_vector_type(4)))  unsigned int  u32x4;
typedef __attribute__((ext_vector_type(4)))  int           i32x4;
typedef __attribute__((ext_vector_type(8)))  int           i32x8;

#if defined(__has_builtin)
#if __has_builtin(__builtin_amdgcn_tensor_load_to_lds) && __has_builtin(__builtin_amdgcn_s_wait_tensorcnt)
#define HAVE_TDM 1
#endif
#endif
#ifndef HAVE_TDM
#define HAVE_TDM 0
#endif

static __device__ __forceinline__ f32x8 wmma_bf16(u32x8 a, u32x8 b, f32x8 c) {
    return __builtin_amdgcn_wmma_f32_16x16x32_bf16(
        /*neg_a=*/false, __builtin_bit_cast(bf16x16, a),
        /*neg_b=*/false, __builtin_bit_cast(bf16x16, b),
        /*c_mod=*/(short)0, c, /*reuse_a=*/false, /*reuse_b=*/false);
}
static __device__ __forceinline__ unsigned short f2bf(float f) {
    unsigned int u = __builtin_bit_cast(unsigned int, f);
    u += 0x7FFFu + ((u >> 16) & 1u);                 // RNE
    return (unsigned short)(u >> 16);
}
static __device__ __forceinline__ unsigned int packbf(float a, float b) {
    return (unsigned int)f2bf(a) | ((unsigned int)f2bf(b) << 16);
}
// 16-bit A-fragment K offset for (vgpr j, lane-half): K = {0..7,16..23} + 8*half
static __device__ __forceinline__ int a_koff(int j, int half) {
    return (j < 4 ? 2 * j : 16 + 2 * (j - 4)) + half * 8;
}

// ---------- problem geometry ----------
#define NTOK    196     // tokens per image (14*14)
#define DMODEL  384
#define NHEADS  16
#define HD      6144    // 16*384
#define ROWS    784     // 4*196
#define MT      49      // ROWS/16
#define KT_D    12      // 384/32
#define NT_HD   384     // 6144/16
#define BATCHES 1536    // 4*384 attention batches
#define NT_S    13      // ceil(196/16)  (S col tiles / Q row tiles, pad to 208)
#define KT_S    7       // ceil(196/32)  (P/V K tiles, pad to 224)
#define KT_HD   192     // 6144/32
#define NT_D    24      // 384/16

// ---------- kernels ----------
__global__ void k_zero(unsigned int* __restrict__ p, int n) {
    int i = blockIdx.x * 256 + threadIdx.x;
    if (i < n) p[i] = 0u;
}

// fp32 W (Kdim x N, row-major) -> bf16 B-fragments [(kt*Ntiles+nt)*32+lane]*8+j
__global__ void k_pack_B(const float* __restrict__ W, unsigned int* __restrict__ out,
                         int Ktiles, int Ntiles, int N) {
    int idx = blockIdx.x * 256 + threadIdx.x;
    int j = idx & 7, l = (idx >> 3) & 31, tile = idx >> 8;
    int nt = tile % Ntiles, kt = tile / Ntiles;
    int n = nt * 16 + (l & 15);
    int krow = kt * 32 + a_koff(j, l >> 4);          // B layout: K = {0..7,16..23} + 8*half
    out[idx] = packbf(W[(size_t)krow * N + n], W[(size_t)(krow + 1) * N + n]);
}

// LayerNorm over 384, one wave32 per row
__global__ void __launch_bounds__(128) k_ln(const float* __restrict__ x,
                                            const float* __restrict__ gamma,
                                            const float* __restrict__ beta,
                                            float* __restrict__ xnf) {
    int lane = threadIdx.x & 31;
    int row  = blockIdx.x * 4 + (threadIdx.x >> 5);
    const float* xr = x + (size_t)row * DMODEL;
    float v[12], s = 0.f, ss = 0.f;
#pragma unroll
    for (int t = 0; t < 12; t++) { float f = xr[lane + 32 * t]; v[t] = f; s += f; ss += f * f; }
#pragma unroll
    for (int m = 16; m >= 1; m >>= 1) { s += __shfl_xor(s, m, 32); ss += __shfl_xor(ss, m, 32); }
    float mu  = s * (1.0f / DMODEL);
    float var = ss * (1.0f / DMODEL) - mu * mu;
    float inv = rsqrtf(var + 1e-5f);
#pragma unroll
    for (int t = 0; t < 12; t++) {
        int c = lane + 32 * t;
        xnf[(size_t)row * DMODEL + c] = (v[t] - mu) * inv * gamma[c] + beta[c];
    }
}

// xn fp32 -> A-fragments [(mt*KT_D+kt)*32+lane]*8+j
__global__ void k_pack_xnA(const float* __restrict__ xnf, unsigned int* __restrict__ out) {
    int idx = blockIdx.x * 256 + threadIdx.x;
    int j = idx & 7, l = (idx >> 3) & 31, tile = idx >> 8;
    int kt = tile % KT_D, mt = tile / KT_D;
    int m = mt * 16 + (l & 15);
    int k = kt * 32 + a_koff(j, l >> 4);
    const float* r = xnf + (size_t)m * DMODEL + k;
    out[idx] = packbf(r[0], r[1]);
}

// QKV projection: (784x384)@(384x6144), epilogue scatters straight into
// Q A-fragments / K B-fragments / V B-fragments (attention-batch layout).
__global__ void __launch_bounds__(256) k_proj(
        const unsigned int* __restrict__ xnA,
        const unsigned int* __restrict__ wqB,
        const unsigned int* __restrict__ wkB,
        const unsigned int* __restrict__ wvB,
        unsigned short* __restrict__ qA16,
        unsigned short* __restrict__ kB16,
        unsigned short* __restrict__ vB16) {
    int lane = threadIdx.x & 31, wave = threadIdx.x >> 5;
    int ntile = blockIdx.x * 8 + wave;               // 0..383
    int mt = blockIdx.y, z = blockIdx.z;
    const unsigned int* Wt = (z == 0) ? wqB : (z == 1) ? wkB : wvB;
    const unsigned int* ap = xnA + (size_t)mt * KT_D * 256 + lane * 8;
    const unsigned int* bp = Wt  + (size_t)ntile * 256 + lane * 8;
    f32x8 acc = {};
#pragma unroll 4
    for (int kt = 0; kt < KT_D; kt++) {
        u32x8 a = *(const u32x8*)ap;
        u32x8 b = *(const u32x8*)bp;
        __builtin_prefetch(bp + (size_t)NT_HD * 256, 0, 1);
        acc = wmma_bf16(a, b, acc);
        ap += 256; bp += (size_t)NT_HD * 256;
    }
    int half = lane >> 4, nl = lane & 15;
    int head = ntile / NT_D;                         // 24 n-tiles per head
    int channel = (ntile % NT_D) * 16 + nl;
#pragma unroll
    for (int i = 0; i < 8; i++) {
        int r = mt * 16 + i + 8 * half;              // C layout: M = i + 8*(lane/16)
        int bs = r / NTOK, nn = r - bs * NTOK;
        int b = bs * DMODEL + channel;               // attention batch
        unsigned short hv = f2bf(acc[i]);
        if (z == 0) {                                // Q -> A-frag (K = head, pad 32)
            int hp = head >> 3, jp = (head & 7) >> 1, pos = head & 1;
            size_t a16 = (((size_t)(b * NT_S + (nn >> 4)) * 32) + hp * 16 + (nn & 15)) * 16 + jp * 2 + pos;
            qA16[a16] = hv;
        } else if (z == 1) {                         // K -> B-frag (K rows = head, pad 32)
            size_t a16 = (((size_t)(b * NT_S + (nn >> 4)) * 32) + (head >> 3) * 16 + (nn & 15)) * 16 + (head & 7);
            kB16[a16] = hv;
        } else {                                     // V -> B-frag (K rows = token, pad 224)
            int kt2 = nn >> 5, h2 = (nn >> 3) & 1;
            size_t a16 = (((size_t)(b * KT_S + kt2) * 32) + h2 * 16 + head) * 16 + ((nn & 7) + ((nn >> 1) & 8));
            vB16[a16] = hv;
        }
    }
}

// Per (batch, 16-row Q tile): TDM streams V-frags into LDS while we do
// S = QK^T/sqrt(384) via WMMA + softmax in LDS; probs -> d_out; repack P to
// A-frags in LDS; O = P@V via WMMA -> aout A-frags.
__global__ void __launch_bounds__(256) k_attn(
        const unsigned int* __restrict__ qA,
        const unsigned int* __restrict__ kB,
        const unsigned int* __restrict__ vB,
        float* __restrict__ probs,
        unsigned short* __restrict__ aoutA16) {
    __shared__ float S[16][208];
    __shared__ __align__(32) unsigned int Pfrag[KT_S][32][8];
    __shared__ __align__(32) unsigned int Vlds[KT_S * 32 * 8];   // 7168 B
    int lane = threadIdx.x & 31, wave = threadIdx.x >> 5;
    int half = lane >> 4, nl = lane & 15;
    int mt = blockIdx.x, b = blockIdx.y;

#if HAVE_TDM
    if (wave == 0) {
        // Tensor DMA: copy this batch's 7 KB of V B-fragments global->LDS,
        // overlapping the whole S/softmax phase. 1-D tile: 896 x 8B elements.
        unsigned long long ga = (unsigned long long)(vB + (size_t)b * KT_S * 256);
        unsigned int lds = (unsigned int)(unsigned long long)(void*)&Vlds[0];
        u32x4 g0;
        g0[0] = 1u;                                       // count=1, user D#
        g0[1] = lds;                                      // lds_addr
        g0[2] = (unsigned int)ga;                         // global_addr[31:0]
        g0[3] = (unsigned int)((ga >> 32) & 0x01FFFFFFu)  // global_addr[56:32]
              | (2u << 30);                               // type=2 ("image")
        i32x8 g1 = {};
        g1[0] = (3 << 16);                                // data_size=8B
        g1[1] = (int)(896u << 16);                        // tensor_dim0[15:0]
        g1[2] = (int)(0u | (1u << 16));                   // td0[31:16]=0, tensor_dim1=1
        g1[3] = (int)(896u << 16);                        // td1[31:16]=0, tile_dim0=896
        g1[4] = 0;                                        // tile_dim1=0, tile_dim2=0
        g1[5] = 896;                                      // tensor_dim0_stride[31:0]
        g1[6] = 0; g1[7] = 0;
        i32x4 gz = {};
#if defined(__clang_major__) && __clang_major__ >= 23
        i32x8 gz8 = {};
        __builtin_amdgcn_tensor_load_to_lds(g0, g1, gz, gz, gz8, 0);
#else
        __builtin_amdgcn_tensor_load_to_lds(g0, g1, gz, gz, 0);
#endif
    }
#endif

    u32x8 qa = *(const u32x8*)(qA + ((size_t)(b * NT_S + mt) * 32 + lane) * 8);
    const float scale = 0.05103103630798288f;        // 1/sqrt(384)
    for (int nt = wave; nt < NT_S; nt += 8) {        // wave-uniform: EXEC stays full
        u32x8 kb = *(const u32x8*)(kB + ((size_t)(b * NT_S + nt) * 32 + lane) * 8);
        f32x8 c = {};
        c = wmma_bf16(qa, kb, c);
#pragma unroll
        for (int i = 0; i < 8; i++) S[i + 8 * half][nt * 16 + nl] = c[i] * scale;
    }
    __syncthreads();
#pragma unroll
    for (int rr = 0; rr < 2; rr++) {                 // 2 rows per wave
        int row = wave * 2 + rr;
        float m = -3.0e38f;
        for (int col = lane; col < NTOK; col += 32) m = fmaxf(m, S[row][col]);
#pragma unroll
        for (int mk = 16; mk >= 1; mk >>= 1) m = fmaxf(m, __shfl_xor(m, mk, 32));
        float s = 0.f;
        for (int col = lane; col < NTOK; col += 32) {
            float e = __expf(S[row][col] - m); S[row][col] = e; s += e;
        }
#pragma unroll
        for (int mk = 16; mk >= 1; mk >>= 1) s += __shfl_xor(s, mk, 32);
        float inv = 1.0f / s;
        int grow = mt * 16 + row;
        for (int col = lane; col < 208; col += 32) {
            if (col < NTOK) {
                float p = S[row][col] * inv; S[row][col] = p;
                if (grow < NTOK) probs[((size_t)(b * NTOK + grow)) * NTOK + col] = p;
            } else S[row][col] = 0.f;                // pad cols for P packing
        }
    }
    __syncthreads();
    if (wave < KT_S) {                               // repack P rows into A-fragments
#pragma unroll
        for (int j = 0; j < 8; j++) {
            int k0 = wave * 32 + a_koff(j, half);
            float v0 = (k0     < 208) ? S[nl][k0]     : 0.f;
            float v1 = (k0 + 1 < 208) ? S[nl][k0 + 1] : 0.f;
            Pfrag[wave][lane][j] = packbf(v0, v1);
        }
    }
    __syncthreads();
    if (wave == 0) {                                 // O = P @ V  (16x16 out tile)
#if HAVE_TDM
        __builtin_amdgcn_s_wait_tensorcnt(0);        // V-frags resident in LDS
#endif
        f32x8 acc = {};
#pragma unroll
        for (int kt = 0; kt < KT_S; kt++) {
            u32x8 pa = *(const u32x8*)&Pfrag[kt][lane][0];
#if HAVE_TDM
            u32x8 vv = *(const u32x8*)&Vlds[(kt * 32 + lane) * 8];
#else
            u32x8 vv = *(const u32x8*)(vB + ((size_t)(b * KT_S + kt) * 32 + lane) * 8);
#endif
            acc = wmma_bf16(pa, vv, acc);
        }
        int bs = b / DMODEL, channel = b - bs * DMODEL;
#pragma unroll
        for (int i = 0; i < 8; i++) {
            int grow = mt * 16 + i + 8 * half;
            if (grow < NTOK) {
                int r = bs * NTOK + grow;
                int k = nl * DMODEL + channel;       // col head = nl
                int kt2 = k >> 5, kk = k & 31;
                int h2 = (kk >> 3) & 1, koff = kk - 8 * h2;
                int jp = (koff < 8) ? (koff >> 1) : 4 + ((koff - 16) >> 1);
                int pos = koff & 1;
                size_t a16 = (((size_t)((r >> 4) * KT_HD + kt2) * 32) + h2 * 16 + (r & 15)) * 16 + jp * 2 + pos;
                aoutA16[a16] = f2bf(acc[i]);
            }
        }
    }
}

// out = aout(784x6144) @ Wout(6144x384) + xn  -> d_out (fp32)
__global__ void __launch_bounds__(256) k_outproj(
        const unsigned int* __restrict__ aoutA,
        const unsigned int* __restrict__ woB,
        const float* __restrict__ xnf,
        float* __restrict__ out) {
    int lane = threadIdx.x & 31, wave = threadIdx.x >> 5;
    int ntile = blockIdx.x * 8 + wave;               // 0..23
    int mt = blockIdx.y;
    const unsigned int* ap = aoutA + (size_t)mt * KT_HD * 256 + lane * 8;
    const unsigned int* bp = woB + (size_t)ntile * 256 + lane * 8;
    f32x8 acc = {};
#pragma unroll 4
    for (int kt = 0; kt < KT_HD; kt++) {
        u32x8 a = *(const u32x8*)ap;
        u32x8 w = *(const u32x8*)bp;
        __builtin_prefetch(ap + 256, 0, 1);
        acc = wmma_bf16(a, w, acc);
        ap += 256; bp += (size_t)NT_D * 256;
    }
    int half = lane >> 4, nl = lane & 15;
    int c = ntile * 16 + nl;
#pragma unroll
    for (int i = 0; i < 8; i++) {
        int r = mt * 16 + i + 8 * half;
        out[(size_t)r * DMODEL + c] = acc[i] + xnf[(size_t)r * DMODEL + c];
    }
}

// ---------- host launch ----------
extern "C" void kernel_launch(void* const* d_in, const int* in_sizes, int n_in,
                              void* d_out, int out_size, void* d_ws, size_t ws_size,
                              hipStream_t stream) {
    (void)in_sizes; (void)n_in; (void)out_size; (void)ws_size;
    const float* x     = (const float*)d_in[0];
    const float* Wq    = (const float*)d_in[1];
    const float* Wk    = (const float*)d_in[2];
    const float* Wv    = (const float*)d_in[3];
    const float* Wout  = (const float*)d_in[4];
    const float* gamma = (const float*)d_in[5];
    const float* beta  = (const float*)d_in[6];
    float* outp  = (float*)d_out;
    float* probs = outp + (size_t)ROWS * DMODEL;     // out(301056) then probs

    char* ws = (char*)d_ws;
    const size_t o_xnf = 0;
    const size_t o_xnA = o_xnf + (size_t)ROWS * DMODEL * 4;
    const size_t o_wqB = o_xnA + (size_t)MT * KT_D * 1024;
    const size_t o_wkB = o_wqB + (size_t)KT_D * NT_HD * 1024;
    const size_t o_wvB = o_wkB + (size_t)KT_D * NT_HD * 1024;
    const size_t o_woB = o_wvB + (size_t)KT_D * NT_HD * 1024;
    const size_t o_QA  = o_woB + (size_t)KT_HD * NT_D * 1024;
    const size_t o_KB  = o_QA  + (size_t)BATCHES * NT_S * 1024;
    const size_t o_VB  = o_KB  + (size_t)BATCHES * NT_S * 1024;
    const size_t o_aA  = o_VB  + (size_t)BATCHES * KT_S * 1024;   // total ~82.2 MB

    float*        xnf = (float*)(ws + o_xnf);
    unsigned int* xnA = (unsigned int*)(ws + o_xnA);
    unsigned int* wqB = (unsigned int*)(ws + o_wqB);
    unsigned int* wkB = (unsigned int*)(ws + o_wkB);
    unsigned int* wvB = (unsigned int*)(ws + o_wvB);
    unsigned int* woB = (unsigned int*)(ws + o_woB);
    unsigned int* qAf = (unsigned int*)(ws + o_QA);
    unsigned int* kBf = (unsigned int*)(ws + o_KB);
    unsigned int* vBf = (unsigned int*)(ws + o_VB);
    unsigned int* aAf = (unsigned int*)(ws + o_aA);

    // zero the padded Q/K/V fragment region (scatter epilogue fills the rest)
    int nz = (int)((o_aA - o_QA) / 4);
    k_zero<<<(nz + 255) / 256, 256, 0, stream>>>(qAf, nz);

    // weight swizzles (once per launch — deterministic)
    k_pack_B<<<KT_D * NT_HD, 256, 0, stream>>>(Wq,   wqB, KT_D,  NT_HD, HD);
    k_pack_B<<<KT_D * NT_HD, 256, 0, stream>>>(Wk,   wkB, KT_D,  NT_HD, HD);
    k_pack_B<<<KT_D * NT_HD, 256, 0, stream>>>(Wv,   wvB, KT_D,  NT_HD, HD);
    k_pack_B<<<KT_HD * NT_D, 256, 0, stream>>>(Wout, woB, KT_HD, NT_D,  DMODEL);

    k_ln<<<ROWS / 4, 128, 0, stream>>>(x, gamma, beta, xnf);
    k_pack_xnA<<<MT * KT_D, 256, 0, stream>>>(xnf, xnA);

    k_proj<<<dim3(NT_HD / 8, MT, 3), 256, 0, stream>>>(
        xnA, wqB, wkB, wvB,
        (unsigned short*)qAf, (unsigned short*)kBf, (unsigned short*)vBf);

    k_attn<<<dim3(NT_S, BATCHES), 256, 0, stream>>>(
        qAf, kBf, vBf, probs, (unsigned short*)aAf);

    k_outproj<<<dim3(NT_D / 8, MT), 256, 0, stream>>>(aAf, woB, xnf, outp);
}
